// InfoNCELoss_19396072308888
// MI455X (gfx1250) — matrix operand order repaired and checked by reading, not compiled, not run on hardware
//
#include <hip/hip_runtime.h>
#include <hip/hip_bf16.h>

// ---------------------------------------------------------------------------
// Fused InfoNCE loss for MI455X (gfx1250, wave32, WMMA).
//   sim = (E E^T)/T computed tile-wise with v_wmma_f32_16x16x32_f16,
//   exp/masking/row-reductions fused (sim never materialized).
//   v2: software-pipelined K-loop (overlap ds_load with WMMA) and
//       branch-free epilogue using raw v_exp_f32 (inputs are L2-normalized,
//       so |sim/T*log2e| <= 28.9 -- no range fixup needed).
// ---------------------------------------------------------------------------

typedef _Float16 h4  __attribute__((ext_vector_type(4)));
typedef _Float16 h8  __attribute__((ext_vector_type(8)));
typedef _Float16 v16h __attribute__((ext_vector_type(16)));
typedef float    v8f  __attribute__((ext_vector_type(8)));

#define Bn 8192
#define Dk 1024

static constexpr float INV_T_LOG2E = 20.0f * 1.4426950408889634f; // (1/0.05)*log2(e)

static constexpr int ROWPAD         = 8;            // halfs of pad -> breaks LDS bank stride
static constexpr int RSTR           = Dk + ROWPAD;  // padded row stride in halfs
static constexpr int ROWS_PER_BLOCK = 128;          // 8 waves x 16 rows
static constexpr int JCHUNK         = 2048;         // columns handled per blockIdx.y
static constexpr size_t SMEM_BYTES  =
    (size_t)(ROWS_PER_BLOCK + 16) * RSTR * sizeof(_Float16);   // 297,216 B < 320 KB WGP LDS

// ---------------------------------------------------------------------------
// Kernel 1: f32 -> f16 conversion of embeddings (L2-normalized, |x| <= 1).
// ---------------------------------------------------------------------------
__global__ void infonce_convert_f16(const float* __restrict__ in,
                                    _Float16* __restrict__ outh) {
  size_t i = ((size_t)blockIdx.x * blockDim.x + threadIdx.x) * 4;
  if (i + 3 < (size_t)Bn * Dk) {
    float4 f = *(const float4*)(in + i);
    h4 o;
    o[0] = (_Float16)f.x; o[1] = (_Float16)f.y;
    o[2] = (_Float16)f.z; o[3] = (_Float16)f.w;
    *(h4*)(outh + i) = o;
  }
}

// ---------------------------------------------------------------------------
// Kernel 2: zero per-row accumulators (graph-replay safe: re-zeroed each call).
// ---------------------------------------------------------------------------
__global__ void infonce_init(float* gSum, float* gNum, int* gHas) {
  int i = blockIdx.x * blockDim.x + threadIdx.x;
  if (i < Bn) { gSum[i] = 0.0f; gNum[i] = 0.0f; gHas[i] = 0; }
}

// ---------------------------------------------------------------------------
// Kernel 3: fused tiled sim + exp + masked row reductions.
//   grid = (Bn/128, Bn/JCHUNK), block = 256 threads (8 wave32 waves).
//   Wave w owns rows [rowBase + 16w, +16); sweeps a 2048-column chunk.
// ---------------------------------------------------------------------------
__global__ void __launch_bounds__(256, 1)
infonce_sim_kernel(const _Float16* __restrict__ Eh,
                   const int* __restrict__ rel,
                   const unsigned char* __restrict__ ist,
                   float* __restrict__ gSum,
                   float* __restrict__ gNum,
                   int* __restrict__ gHas) {
  extern __shared__ _Float16 smem[];
  _Float16* As = smem;                              // [128][RSTR] A slab
  _Float16* Bs = smem + ROWS_PER_BLOCK * RSTR;      // [16][RSTR]  B tile slab

  const int tid  = threadIdx.x;
  const int wave = tid >> 5;
  const int lane = tid & 31;
  const int l15  = lane & 15;
  const int hsel = lane >> 4;                       // 0: lanes 0-15, 1: lanes 16-31

  const int rowBase = blockIdx.x * ROWS_PER_BLOCK;
  const int jStart  = blockIdx.y * JCHUNK;
  const int jEnd    = jStart + JCHUNK;

  // ---- stage A slab: 128 rows x 1024 halfs (256 KB) via b128 chunks ----
  for (int idx = tid; idx < ROWS_PER_BLOCK * (Dk / 8); idx += 256) {
    int r = idx >> 7;                               // 128 chunks per row
    int c = idx & 127;
    *(h8*)(As + r * RSTR + c * 8) =
        *(const h8*)(Eh + (size_t)(rowBase + r) * Dk + c * 8);
  }
  __syncthreads();

  // Per-lane constant row metadata: this lane accumulates 8 fixed rows
  // (C-matrix layout: lane half selects M in {v} or {8+v}, VGPR index = v).
  const int myRow0 = rowBase + wave * 16 + hsel * 8;
  int relI[8]; int istI[8];
#pragma unroll
  for (int v = 0; v < 8; ++v) {
    relI[v] = rel[myRow0 + v];
    istI[v] = (int)ist[myRow0 + v];
  }

  float accS[8], accN[8]; int accH[8];
#pragma unroll
  for (int v = 0; v < 8; ++v) { accS[v] = 0.0f; accN[v] = 0.0f; accH[v] = 0; }

  // A 16x32 f16 frag: lanes 0-15 hold K {k0..k0+7, k0+16..k0+23},
  // lanes 16-31 hold K {k0+8..k0+15, k0+24..k0+31}.
  // B 32x16 f16 frag: same K split as A (lanes 0-15 {k0..k0+7, k0+16..k0+23}, lanes 16-31 {k0+8..k0+15, k0+24..k0+31}).
  const int kAoff = hsel * 8;
  const int kBoff = hsel * 8;
  const _Float16* Arow = As + (wave * 16 + l15) * RSTR;  // A-frag row for this lane
  const _Float16* Brow = Bs + l15 * RSTR;                // B-frag column (= E row)

  for (int jt = jStart; jt < jEnd; jt += 16) {
    // ---- stage B tile slab: 16 rows x 1024 halfs (32 KB) ----
    __syncthreads();
    for (int idx = tid; idx < 16 * (Dk / 8); idx += 256) {
      int r = idx >> 7;
      int c = idx & 127;
      *(h8*)(Bs + r * RSTR + c * 8) =
          *(const h8*)(Eh + (size_t)(jt + r) * Dk + c * 8);
    }
    __syncthreads();

    // Prefetch next tile into cache hierarchy (global_prefetch_b8 path).
    if (jt + 16 < jEnd)
      __builtin_prefetch((const char*)(Eh + (size_t)(jt + 16) * Dk) + tid * 128, 0, 1);

    const int j    = jt + l15;
    const int relJ = rel[j];
    const int istJ = (int)ist[j];

    // ---- software-pipelined K loop: fragments for step k are loaded in
    //      step k-32, so the WMMA only waits on older ds_loads. ----
    h8 a0 = *(const h8*)(Arow + kAoff);
    h8 a1 = *(const h8*)(Arow + kAoff + 16);
    h8 b0 = *(const h8*)(Brow + kBoff);
    h8 b1 = *(const h8*)(Brow + kBoff + 16);

    v8f c = {};
#pragma unroll 4
    for (int k0 = 0; k0 < Dk; k0 += 32) {
      const int kn = (k0 + 32) & (Dk - 1);          // wraps to 0 on last iter
      h8 na0 = *(const h8*)(Arow + kn + kAoff);
      h8 na1 = *(const h8*)(Arow + kn + kAoff + 16);
      h8 nb0 = *(const h8*)(Brow + kn + kBoff);
      h8 nb1 = *(const h8*)(Brow + kn + kBoff + 16);

      v16h a, b;
#pragma unroll
      for (int t = 0; t < 8; ++t) {
        a[t] = a0[t]; a[8 + t] = a1[t];
        b[t] = b0[t]; b[8 + t] = b1[t];
      }
      c = __builtin_amdgcn_wmma_f32_16x16x32_f16(
          /*neg_a=*/false, a, /*neg_b=*/false, b,
          /*c_mod=*/(short)0, c, /*reuse_a=*/false, /*reuse_b=*/false);

      a0 = na0; a1 = na1; b0 = nb0; b1 = nb1;
    }

    // Branch-free fused epilogue: raw v_exp_f32 (range-safe: |arg| <= 28.9),
    // diagonal + positive masks via cndmask selects.
#pragma unroll
    for (int v = 0; v < 8; ++v) {
      const int i = myRow0 + v;
      float e = __builtin_amdgcn_exp2f(c[v] * INV_T_LOG2E);  // exp(sim/T)
      e = (i != j) ? e : 0.0f;                               // diag -> 0
      const bool pos = (i != j) & (istI[v] & istJ) & (relI[v] == relJ);
      accS[v] += e;
      accN[v] += pos ? e : 0.0f;
      accH[v] |= pos ? 1 : 0;
    }
  }

  // Reduce over the 16 columns held by the 16-lane half (xor stays in-half).
#pragma unroll
  for (int off = 1; off < 16; off <<= 1) {
#pragma unroll
    for (int v = 0; v < 8; ++v) {
      accS[v] += __shfl_xor(accS[v], off, 32);
      accN[v] += __shfl_xor(accN[v], off, 32);
      accH[v] |= __shfl_xor(accH[v], off, 32);
    }
  }

  if (l15 == 0) {                                   // lanes 0 and 16 publish
#pragma unroll
    for (int v = 0; v < 8; ++v) {
      atomicAdd(&gSum[myRow0 + v], accS[v]);
      atomicAdd(&gNum[myRow0 + v], accN[v]);
      atomicOr(&gHas[myRow0 + v], accH[v]);
    }
  }
}

// ---------------------------------------------------------------------------
// Kernel 4: per-row loss, masked mean, positive-row count.
// ---------------------------------------------------------------------------
__global__ void infonce_finalize(const float* __restrict__ gSum,
                                 const float* __restrict__ gNum,
                                 const int* __restrict__ gHas,
                                 float* __restrict__ out) {
  __shared__ float sL[256];
  __shared__ int   sN[256];
  float L = 0.0f; int n = 0;
  for (int i = threadIdx.x; i < Bn; i += 256) {
    const int h = gHas[i];
    const float num = h ? gNum[i] : 1.0f;
    const float l = -logf(num / (gSum[i] + 1e-8f));
    if (h) { L += l; n += 1; }
  }
  sL[threadIdx.x] = L; sN[threadIdx.x] = n;
  __syncthreads();
  for (int s = 128; s > 0; s >>= 1) {
    if (threadIdx.x < s) {
      sL[threadIdx.x] += sL[threadIdx.x + s];
      sN[threadIdx.x] += sN[threadIdx.x + s];
    }
    __syncthreads();
  }
  if (threadIdx.x == 0) {
    const int nr = sN[0];
    out[0] = (nr > 0) ? (sL[0] / (float)nr) : 0.0f;  // loss
    out[1] = (float)nr;                               // num_positives
  }
}

// ---------------------------------------------------------------------------
extern "C" void kernel_launch(void* const* d_in, const int* in_sizes, int n_in,
                              void* d_out, int out_size, void* d_ws, size_t ws_size,
                              hipStream_t stream) {
  (void)in_sizes; (void)n_in; (void)out_size; (void)ws_size;

  const float*         emb = (const float*)d_in[0];
  const int*           rel = (const int*)d_in[1];
  const unsigned char* ist = (const unsigned char*)d_in[2];
  float*               out = (float*)d_out;

  char* ws = (char*)d_ws;
  _Float16* Eh  = (_Float16*)ws;                                    // 16 MB
  float*    gS  = (float*)(ws + (size_t)Bn * Dk * sizeof(_Float16));
  float*    gN  = gS + Bn;
  int*      gH  = (int*)(gN + Bn);

  hipFuncSetAttribute(reinterpret_cast<const void*>(infonce_sim_kernel),
                      hipFuncAttributeMaxDynamicSharedMemorySize,
                      (int)SMEM_BYTES);

  infonce_convert_f16<<<(Bn * Dk / 4 + 255) / 256, 256, 0, stream>>>(emb, Eh);
  infonce_init<<<(Bn + 255) / 256, 256, 0, stream>>>(gS, gN, gH);

  dim3 grid(Bn / ROWS_PER_BLOCK, Bn / JCHUNK);   // 64 x 4 = 256 workgroups
  infonce_sim_kernel<<<grid, 256, SMEM_BYTES, stream>>>(Eh, rel, ist, gS, gN, gH);

  infonce_finalize<<<1, 256, 0, stream>>>(gS, gN, gH, out);
}
